// MultiHeadAttention_11553462026349
// MI455X (gfx1250) — compile-verified, matrix operand order checked by reading, not run on hardware
//
#include <hip/hip_runtime.h>
#include <hip/hip_bf16.h>

#define NB 4
#define LQ 2048
#define LK 2048
#define DMODEL 1024
#define NH 16
#define DH 64

typedef __bf16 v16bf __attribute__((ext_vector_type(16)));
typedef __bf16 v8bf  __attribute__((ext_vector_type(8)));
typedef __bf16 v4bf  __attribute__((ext_vector_type(4)));
typedef float  v8f   __attribute__((ext_vector_type(8)));
typedef float  v4f   __attribute__((ext_vector_type(4)));
typedef int    v4i   __attribute__((ext_vector_type(4)));
typedef __attribute__((address_space(1))) v4i* gv4i_p;   // global (AS1)
typedef __attribute__((address_space(3))) v4i* lv4i_p;   // LDS (AS3)

// Build a 16x32 (16-bit) WMMA A/B fragment from a row-major tile whose rows are
// K-contiguous. Layout per CDNA5 ISA 7.12.2: lane r=lane&15 holds row/col r,
// k-group g=lane>>4 -> halves {g*8..g*8+7} and {16+g*8..16+g*8+7}.
__device__ __forceinline__ v16bf make_frag(const __bf16* p) {
    v8bf lo = *(const v8bf*)p;
    v8bf hi = *(const v8bf*)(p + 16);
    v16bf r;
#pragma unroll
    for (int i = 0; i < 8; ++i) { r[i] = lo[i]; r[8 + i] = hi[i]; }
    return r;
}

__device__ __forceinline__ v16bf frag16(const __bf16* tile, int ld) {
    int lane = threadIdx.x & 31;
    return make_frag(tile + (lane & 15) * ld + (lane >> 4) * 8);
}

__device__ __forceinline__ v8f wmma_bf16(v16bf a, v16bf b, v8f c) {
    return __builtin_amdgcn_wmma_f32_16x16x32_bf16(false, a, false, b, (short)0, c,
                                                   false, false);
}

// --- CDNA5 async copy: 16B global -> LDS, tracked by ASYNCcnt -------------
__device__ __forceinline__ void async_copy16(__bf16* lds, const __bf16* g) {
#if __has_builtin(__builtin_amdgcn_global_load_async_to_lds_b128)
    v4i* gp = (v4i*)g;     // drop const, reinterpret (generic)
    v4i* lp = (v4i*)lds;
    __builtin_amdgcn_global_load_async_to_lds_b128((gv4i_p)gp, (lv4i_p)lp, 0, 0);
#else
    unsigned lo = (unsigned)(unsigned long long)
        (__attribute__((address_space(3))) void*)lds;
    asm volatile("global_load_async_to_lds_b128 %0, %1, off"
                 :: "v"(lo), "v"(g) : "memory");
#endif
}

__device__ __forceinline__ void wait_async0() {
#if __has_builtin(__builtin_amdgcn_s_wait_asynccnt)
    __builtin_amdgcn_s_wait_asynccnt(0);
#else
    asm volatile("s_wait_asynccnt 0x0" ::: "memory");
#endif
}

// ---------------------------------------------------------------------------
// Kernel 1: fused Q/K/V projection  C[8192,1024] = X[8192,1024] @ W[1024,1024]^T
// Output bf16; Q,K as [n,h,L,DH]; V transposed as [n,h,DH,L].
// 8 waves; workgroup tile 64(M) x 64(N); k-chunks of 64 staged through LDS
// (float4 loads -> packed bf16 cvt -> ds_store_b64).
// ---------------------------------------------------------------------------
__global__ __launch_bounds__(256) void proj_qkv(
    const float* __restrict__ query, const float* __restrict__ keyx,
    const float* __restrict__ Wq, const float* __restrict__ Wk,
    const float* __restrict__ Wv,
    __bf16* __restrict__ Qb, __bf16* __restrict__ Kb, __bf16* __restrict__ Vt)
{
    __shared__ __align__(16) __bf16 As[64 * 72];   // stride 72: 144B rows (16B mult)
    __shared__ __align__(16) __bf16 Bs[64 * 72];
    const int tid  = threadIdx.x;
    const int wave = tid >> 5;
    const int lane = tid & 31;
    const int z = blockIdx.z;
    const float* __restrict__ X = (z == 0) ? query : keyx;
    const float* __restrict__ W = (z == 0) ? Wq : ((z == 1) ? Wk : Wv);
    const int mbase = blockIdx.x * 64;   // 8192 rows
    const int nbase = blockIdx.y * 64;   // 1024 cols
    const int mt = wave >> 1;            // wave's M sub-tile (0..3)
    const int np = wave & 1;             // wave's N pair (0..1)
    v8f acc0 = {}; v8f acc1 = {};
    for (int kc = 0; kc < DMODEL; kc += 64) {
#pragma unroll
        for (int i = 0; i < 4; ++i) {           // 1024 float4 per matrix
            int idx = i * 256 + tid;
            int row = idx >> 4, c4 = idx & 15;  // col = c4*4
            v4f xa = *(const v4f*)&X[(size_t)(mbase + row) * DMODEL + kc + c4 * 4];
            v4f wb = *(const v4f*)&W[(size_t)(nbase + row) * DMODEL + kc + c4 * 4];
            v4bf xb, wc;
#pragma unroll
            for (int e = 0; e < 4; ++e) { xb[e] = (__bf16)xa[e]; wc[e] = (__bf16)wb[e]; }
            *(v4bf*)&As[row * 72 + c4 * 4] = xb;
            *(v4bf*)&Bs[row * 72 + c4 * 4] = wc;
        }
        __syncthreads();
        v16bf a0 = frag16(As + mt * 16 * 72, 72);
        v16bf a1 = frag16(As + mt * 16 * 72 + 32, 72);
        {
            const __bf16* bt = Bs + (np * 2 + 0) * 16 * 72;
            acc0 = wmma_bf16(a0, frag16(bt, 72), acc0);
            acc0 = wmma_bf16(a1, frag16(bt + 32, 72), acc0);
        }
        {
            const __bf16* bt = Bs + (np * 2 + 1) * 16 * 72;
            acc1 = wmma_bf16(a0, frag16(bt, 72), acc1);
            acc1 = wmma_bf16(a1, frag16(bt + 32, 72), acc1);
        }
        __syncthreads();
    }
    // branchless epilogue: pointer + index select hoisted per block
    const int hi = lane >> 4, nn = lane & 15;
    __bf16* const dstB = (z == 0) ? Qb : ((z == 1) ? Kb : Vt);
    const bool tr = (z == 2);
#pragma unroll
    for (int t = 0; t < 2; ++t) {
        v8f acc = t ? acc1 : acc0;
        int nt = np * 2 + t;
#pragma unroll
        for (int i = 0; i < 8; ++i) {
            int m = mbase + mt * 16 + i + hi * 8;     // global row (n*L + l)
            int c = nbase + nt * 16 + nn;             // global col (h*DH + dh)
            int ns = m >> 11, l = m & (LQ - 1);
            int hh = c >> 6,  dh = c & (DH - 1);
            size_t headoff = (size_t)(ns * NH + hh) * (size_t)(LK * DH);
            size_t idx = tr ? ((size_t)dh * LK + l) : ((size_t)l * DH + dh);
            dstB[headoff + idx] = (__bf16)acc[i];
        }
    }
}

// ---------------------------------------------------------------------------
// Kernel 2: flash attention. One workgroup per (n, h, 128-query block); each of
// the 8 waves owns a 16-query tile. K/V^T tiles (32 keys) double-buffered in
// LDS via GLOBAL_LOAD_ASYNC_TO_LDS_B128 (ASYNCcnt). Online softmax via
// shfl_xor inside 16-lane row groups (matches WMMA f32 C layout).
// ---------------------------------------------------------------------------
__global__ __launch_bounds__(256) void flash_attn(
    const __bf16* __restrict__ Qb, const __bf16* __restrict__ Kb,
    const __bf16* __restrict__ Vt, const unsigned char* __restrict__ pad,
    float* __restrict__ out)
{
    __shared__ __align__(16) __bf16 Ks[2][32 * 64];   // [buf][j][d]
    __shared__ __align__(16) __bf16 Vs[2][64 * 32];   // [buf][d][j]
    __shared__ __align__(16) __bf16 Ps[8][16 * 40];   // per-wave P tile
    __shared__ int s_valid;
    const int tid  = threadIdx.x;
    const int wave = tid >> 5;
    const int lane = tid & 31;
    const int hi = lane >> 4, nn = lane & 15;
    const int n = blockIdx.z, h = blockIdx.y;
    const int qbase = blockIdx.x * 128;

    const size_t headoff = (size_t)(n * NH + h) * (size_t)(LK * DH);
    const __bf16* Kg = Kb + headoff;
    const __bf16* Vg = Vt + headoff;
    // per-thread staging coordinates (16B chunks)
    const int sj = tid >> 3, sch = tid & 7;   // K: row j, 8-elem chunk
    const int sd = tid >> 2, sc2 = tid & 3;   // V^T: row d, 8-elem chunk

    if (tid == 0) s_valid = 0;
    // kick off buffer 0 while we count the padding mask
    async_copy16(&Ks[0][sj * 64 + sch * 8], Kg + (size_t)sj * DH + sch * 8);
    async_copy16(&Vs[0][sd * 32 + sc2 * 8], Vg + (size_t)sd * LK + sc2 * 8);
    __syncthreads();
    {   // recover valid_len[n] from the padding mask (bool bytes)
        const unsigned char* pr = pad + n * LK;
        int c = 0;
#pragma unroll
        for (int i = 0; i < 8; ++i) c += (pr[tid * 8 + i] == 0) ? 1 : 0;
        atomicAdd(&s_valid, c);
    }
    __syncthreads();
    const int valid = s_valid;
    const float scale = rsqrtf((float)valid);   // reference: / sqrt(#valid keys)

    const int mrow0 = qbase + wave * 16;
    const __bf16* qtile = Qb + headoff + (size_t)mrow0 * DH;
    v16bf qa0 = frag16(qtile, DH);        // d = 0..31
    v16bf qa1 = frag16(qtile + 32, DH);   // d = 32..63

    float mrow[8], lsum[8];
#pragma unroll
    for (int i = 0; i < 8; ++i) { mrow[i] = -3.0e38f; lsum[i] = 0.f; }
    v8f o0 = {}, o1 = {}, o2 = {}, o3 = {};

    const int kEnd    = min(valid, qbase + 128);   // block-wide key bound
    const int kEndW   = min(valid, mrow0 + 16);    // this wave's causal bound
    const int nBlocks = (kEnd + 31) >> 5;          // >= 1
    for (int ib = 0; ib < nBlocks; ++ib) {
        const int kb = ib * 32;
        const int buf = ib & 1;
        wait_async0();        // this wave's chunks of buf are in LDS
        __syncthreads();      // everyone's chunks are in LDS; prev reads of buf^1 done
        if (ib + 1 < nBlocks) {   // prefetch next block into the other buffer
            const int kn = kb + 32;
            async_copy16(&Ks[buf ^ 1][sj * 64 + sch * 8],
                         Kg + (size_t)(kn + sj) * DH + sch * 8);
            async_copy16(&Vs[buf ^ 1][sd * 32 + sc2 * 8],
                         Vg + (size_t)sd * LK + kn + sc2 * 8);
        }
        if (kb < kEndW) {   // wave-uniform: EXEC stays all-ones around WMMA
            const __bf16* ks = &Ks[buf][0];
            const __bf16* vs = &Vs[buf][0];
            // S(16x32) = Q(16x64) . K(32x64)^T : two 16x16 tiles, 2 WMMA each
            v8f s0 = {}, s1 = {};
            s0 = wmma_bf16(qa0, frag16(ks, 64), s0);
            s0 = wmma_bf16(qa1, frag16(ks + 32, 64), s0);
            s1 = wmma_bf16(qa0, frag16(ks + 16 * 64, 64), s1);
            s1 = wmma_bf16(qa1, frag16(ks + 16 * 64 + 32, 64), s1);
#pragma unroll
            for (int i = 0; i < 8; ++i) {
                int m  = mrow0 + i + hi * 8;
                int j0 = kb + nn, j1 = kb + 16 + nn;
                float v0 = s0[i] * scale; if (j0 > m || j0 >= valid) v0 = -1.0e30f;
                float v1 = s1[i] * scale; if (j1 > m || j1 >= valid) v1 = -1.0e30f;
                float mx = fmaxf(v0, v1);
                mx = fmaxf(mx, __shfl_xor(mx, 1));
                mx = fmaxf(mx, __shfl_xor(mx, 2));
                mx = fmaxf(mx, __shfl_xor(mx, 4));
                mx = fmaxf(mx, __shfl_xor(mx, 8));
                float mnew  = fmaxf(mrow[i], mx);
                float alpha = __expf(mrow[i] - mnew);
                mrow[i] = mnew;
                float p0 = __expf(v0 - mnew);
                float p1 = __expf(v1 - mnew);
                float rs = p0 + p1;
                rs += __shfl_xor(rs, 1);
                rs += __shfl_xor(rs, 2);
                rs += __shfl_xor(rs, 4);
                rs += __shfl_xor(rs, 8);
                lsum[i] = lsum[i] * alpha + rs;
                o0[i] *= alpha; o1[i] *= alpha; o2[i] *= alpha; o3[i] *= alpha;
                // C-layout -> A-layout crossover goes through LDS
                Ps[wave][(i + hi * 8) * 40 + nn]      = (__bf16)p0;
                Ps[wave][(i + hi * 8) * 40 + 16 + nn] = (__bf16)p1;
            }
            // same-wave DS ops are in-order; compiler inserts the dscnt wait
            v16bf pa = frag16(&Ps[wave][0], 40);    // P as 16x32 A-fragment
            o0 = wmma_bf16(pa, frag16(vs, 32), o0);             // d  0..15
            o1 = wmma_bf16(pa, frag16(vs + 16 * 32, 32), o1);   // d 16..31
            o2 = wmma_bf16(pa, frag16(vs + 32 * 32, 32), o2);   // d 32..47
            o3 = wmma_bf16(pa, frag16(vs + 48 * 32, 32), o3);   // d 48..63
        }
    }
#pragma unroll
    for (int i = 0; i < 8; ++i) {
        float inv = 1.0f / lsum[i];
        int m = mrow0 + i + hi * 8;
        float* orow = out + (size_t)(n * LQ + m) * DMODEL + h * DH + nn;
        orow[0]  = o0[i] * inv;
        orow[16] = o1[i] * inv;
        orow[32] = o2[i] * inv;
        orow[48] = o3[i] * inv;
    }
}

extern "C" void kernel_launch(void* const* d_in, const int* in_sizes, int n_in,
                              void* d_out, int out_size, void* d_ws, size_t ws_size,
                              hipStream_t stream) {
    const float* query = (const float*)d_in[0];
    const float* keyx  = (const float*)d_in[1];
    const float* Wq    = (const float*)d_in[2];
    const float* Wk    = (const float*)d_in[3];
    const float* Wv    = (const float*)d_in[4];
    // d_in[5] = causal mask: recomputed analytically on device (j > i)
    const unsigned char* pad = (const unsigned char*)d_in[6];

    const size_t elems = (size_t)NB * NH * (size_t)LK * DH;  // 8Mi bf16 each
    __bf16* Qb = (__bf16*)d_ws;
    __bf16* Kb = Qb + elems;
    __bf16* Vt = Kb + elems;   // total 48 MB of workspace

    proj_qkv<<<dim3((NB * LQ) / 64, DMODEL / 64, 3), 256, 0, stream>>>(
        query, keyx, Wq, Wk, Wv, Qb, Kb, Vt);
    flash_attn<<<dim3(LQ / 128, NH, NB), 256, 0, stream>>>(
        Qb, Kb, Vt, pad, (float*)d_out);
}
